// BasicBlock_1w8a_q_20426864460062
// MI455X (gfx1250) — matrix operand order repaired and checked by reading, not compiled, not run on hardware
//
#include <hip/hip_runtime.h>
#include <stdint.h>

typedef __attribute__((ext_vector_type(16))) _Float16 v16h;
typedef __attribute__((ext_vector_type(8)))  _Float16 h8;
typedef __attribute__((ext_vector_type(4)))  _Float16 h4;
typedef __attribute__((ext_vector_type(8)))  float    v8f;
typedef __attribute__((ext_vector_type(4)))  int      v4i;

#define CC   128      // channels
#define HH   56
#define WW2  56
#define HW   3136     // 56*56
#define NPIX 100352   // 32*3136
#define KK   1152     // 128*9
#define KC   32       // WMMA K per step (f16)
#define KT   36       // 1152/32
#define LDA  40       // LDS row stride in halves (80B: 16B-aligned frags, bank-spread)

// ---- CDNA5 async global->LDS staging (guarded; falls back to register pipeline) ----
#if defined(__AMDGCN__) && __has_builtin(__builtin_amdgcn_global_load_async_to_lds_b128)
#define HAVE_ASYNC 1
#else
#define HAVE_ASYNC 0
#endif

#if HAVE_ASYNC
#if __has_builtin(__builtin_amdgcn_s_wait_asynccnt)
#define WAIT_ASYNC() __builtin_amdgcn_s_wait_asynccnt(0)
#else
#define WAIT_ASYNC() asm volatile("s_wait_asynccnt 0" ::: "memory")
#endif
// 16B per lane, global -> LDS, tracked by ASYNCcnt.
// Probe-derived signature: (v4i* src_global, AS3 v4i* dst_lds, imm offset, imm cpol)
#define ASYNC_CP16(dst, src)                                                      \
    __builtin_amdgcn_global_load_async_to_lds_b128(                               \
        (v4i*)(src),                                                              \
        (__attribute__((address_space(3))) v4i*)(uintptr_t)(dst), 0, 0)
#endif

// ---------------- kernel 1: pack sign(w) -> f16 [layer][O][k], k=(r*3+s)*128+c ; sw[o]=mean|w| ----
__global__ __launch_bounds__(128) void pack_w_kernel(const float* __restrict__ w1,
                                                     const float* __restrict__ w2,
                                                     _Float16* __restrict__ wp,   // [2][128][1152]
                                                     float* __restrict__ sw) {    // [2][128]
    __shared__ float red[128];
    const int o = blockIdx.x & 127;
    const int l = blockIdx.x >> 7;
    const float* W = (l ? w2 : w1) + (size_t)o * KK;   // [c][r][s] within o
    _Float16* dst = wp + ((size_t)l * CC + o) * KK;
    float acc = 0.0f;
    for (int i = threadIdx.x; i < KK; i += 128) {
        float v = W[i];
        acc += fabsf(v);
        int c = i / 9, rs = i % 9;
        float s = (v > 0.0f) ? 1.0f : ((v < 0.0f) ? -1.0f : 0.0f);
        dst[rs * CC + c] = (_Float16)s;
    }
    red[threadIdx.x] = acc;
    __syncthreads();
    for (int st = 64; st > 0; st >>= 1) {
        if (threadIdx.x < st) red[threadIdx.x] += red[threadIdx.x + st];
        __syncthreads();
    }
    if (threadIdx.x == 0) sw[l * CC + o] = red[0] / (float)KK;
}

// ---------------- kernel 2: folded-BN quantization constants (+ zero pad init) -------------------
__global__ __launch_bounds__(128) void bn_prep_kernel(const float* __restrict__ g1, const float* __restrict__ b1,
                                                      const float* __restrict__ m1, const float* __restrict__ v1,
                                                      const float* __restrict__ g2, const float* __restrict__ b2,
                                                      const float* __restrict__ m2, const float* __restrict__ v2,
                                                      float* __restrict__ bw,   // [2][128]
                                                      float* __restrict__ bb,   // [2][128]
                                                      float* __restrict__ scal, // [2][2]: sc, Th per layer
                                                      float* __restrict__ zpad) // 32 floats = 128B of zeros
{
    __shared__ float red[128];
    const int c = threadIdx.x;
    if (c < 32) zpad[c] = 0.0f;
    for (int l = 0; l < 2; ++l) {
        const float* G = l ? g2 : g1;  const float* B = l ? b2 : b1;
        const float* M = l ? m2 : m1;  const float* V = l ? v2 : v1;
        float wv = G[c] / sqrtf(V[c] + 1e-5f);
        float bv = B[c] - wv * M[c];
        red[c] = fabsf(wv); __syncthreads();
        for (int st = 64; st > 0; st >>= 1) { if (c < st) red[c] = fmaxf(red[c], red[c + st]); __syncthreads(); }
        float Tw = red[0]; __syncthreads();
        red[c] = fabsf(bv); __syncthreads();
        for (int st = 64; st > 0; st >>= 1) { if (c < st) red[c] = fmaxf(red[c], red[c + st]); __syncthreads(); }
        float Tb = fminf(fmaxf(red[0], 1e-10f), 255.0f); __syncthreads();

        const float Ta = 7.0f * 3.0f * 3.0f * 64.0f;  // 4032
        float bwv = rintf(fminf(fmaxf(wv, -Tw), Tw) / Tw * 7.0f);                      // 3-bit levels
        float qb  = rintf(fminf(fmaxf(bv, -Tb), Tb) / Tb * 16383.0f) / 16383.0f * Tb;  // 14-bit quant
        float bbv = rintf(qb * 7.0f * 1023.0f / Ta * 7.0f / Tw);
        bw[l * CC + c] = bwv;
        bb[l * CC + c] = bbv;
        if (c == 0) {
            scal[l * 2 + 0] = rintf(1023.0f / Ta * 7.0f / Tw);          // sc
            scal[l * 2 + 1] = rintf(7.0f * 1023.0f / Ta * 7.0f / Tw);   // Th
        }
        __syncthreads();
    }
}

// ---------------- kernel 3: x NCHW f32 -> NHWC f16 (4 channels per lane, 8B stores) --------------
__global__ __launch_bounds__(256) void to_nhwc_kernel(const float* __restrict__ x, _Float16* __restrict__ xT) {
    size_t idx = (size_t)blockIdx.x * 256 + threadIdx.x;      // (c/4)-major: coalesced reads over p
    int cg = (int)(idx / NPIX);     // 0..31 -> channels cg*4 .. cg*4+3
    int p  = (int)(idx % NPIX);
    int n = p / HW, hw = p % HW;
    const float* src = x + (size_t)n * CC * HW + (size_t)cg * 4 * HW + hw;
    h4 v;
    v.x = (_Float16)src[0];
    v.y = (_Float16)src[HW];
    v.z = (_Float16)src[2 * HW];
    v.w = (_Float16)src[3 * HW];
    *(h4*)(xT + (size_t)p * CC + cg * 4) = v;
}

// ---------------- kernel 4: implicit-GEMM conv + quantized epilogue ------------------------------
// M tile = 128 pixels, N = 128 out-channels, K = 1152 in 36 steps of 32 (fully unrolled),
// double-buffered LDS; staging via GLOBAL_LOAD_ASYNC_TO_LDS_B128 (ASYNCcnt) when available.
__global__ __launch_bounds__(256) void conv_wmma_kernel(const _Float16* __restrict__ actT,    // NHWC f16
                                                        const _Float16* __restrict__ wPackT,  // [128][1152] f16
                                                        const float* __restrict__ sw,
                                                        const float* __restrict__ bw,
                                                        const float* __restrict__ bb,
                                                        const float* __restrict__ scal,       // [sc, Th]
                                                        const _Float16* __restrict__ zpad,    // 64+ halves of zeros
                                                        const float* __restrict__ resid_nchw, // layer0: x
                                                        const _Float16* __restrict__ resid_nhwc, // layer1: x1T
                                                        _Float16* __restrict__ outT,          // layer0: x1T
                                                        float* __restrict__ out_nchw,         // layer1: d_out
                                                        int layer)
{
    __shared__ __align__(16) _Float16 As[2][128 * LDA];
    __shared__ __align__(16) _Float16 Bs[2][128 * LDA];

    const int tid = threadIdx.x;
    // staging role: each thread stages 16 halves (32B) of A and of B per K-chunk
    const int srow = tid >> 1;          // 0..127
    const int shalf = (tid & 1) * 16;   // half offset within 32-half row
    const int sp = blockIdx.x * 128 + srow;
    const int sn = sp / HW, shw = sp % HW;
    const int sh = shw / WW2, sw_ = shw % WW2;
    const _Float16* bSrcRow = wPackT + (size_t)srow * KK + shalf;
    const _Float16* zSrc = zpad + shalf;

    // wave/lane fragment roles (ISA 7.12.2 layouts)
    const int wv = tid >> 5, lane = tid & 31;
    const int l15 = lane & 15, lhi = lane >> 4;
    const int am = wv * 16 + l15;       // A fragment row
    const int akh = lhi * 8;            // A fragment K-half base
    const int bks = lhi * 8;            // B fragment K base

    v8f acc[8];
    #pragma unroll
    for (int i = 0; i < 8; ++i) acc[i] = (v8f)0.0f;

    // A source pointer for chunk kn: branch-free zero-pad select
    auto aSrc = [&](int kn) -> const _Float16* {
        const int rs = kn >> 2;               // tap 0..8 (constant after unroll)
        const int r = rs / 3, s = rs % 3;
        const int c0 = (kn & 3) * KC;
        const int hs = sh + r - 1, wsx = sw_ + s - 1;
        const bool ok = (hs >= 0) & (hs < HH) & (wsx >= 0) & (wsx < WW2);
        const _Float16* p = actT + ((size_t)(sn * HW + hs * WW2 + wsx) * CC + c0 + shalf);
        return ok ? p : zSrc;
    };

#if HAVE_ASYNC
    auto issue = [&](int kn) {   // 4 async 16B transfers into buffer kn&1
        const _Float16* sa = aSrc(kn);
        _Float16* da = &As[kn & 1][srow * LDA + shalf];
        ASYNC_CP16(da,     sa);
        ASYNC_CP16(da + 8, sa + 8);
        const _Float16* sb = bSrcRow + kn * KC;
        _Float16* db = &Bs[kn & 1][srow * LDA + shalf];
        ASYNC_CP16(db,     sb);
        ASYNC_CP16(db + 8, sb + 8);
    };
    issue(0);
    WAIT_ASYNC();
    __syncthreads();
#else
    auto fetch = [&](int kn, h8& a0, h8& a1, h8& b0, h8& b1) {
        const h8* sa = (const h8*)aSrc(kn);
        a0 = sa[0]; a1 = sa[1];
        const h8* sb = (const h8*)(bSrcRow + kn * KC);
        b0 = sb[0]; b1 = sb[1];
    };
    auto stage = [&](int buf, const h8& a0, const h8& a1, const h8& b0, const h8& b1) {
        h8* dA = (h8*)(&As[buf][srow * LDA + shalf]);
        dA[0] = a0; dA[1] = a1;
        h8* dB = (h8*)(&Bs[buf][srow * LDA + shalf]);
        dB[0] = b0; dB[1] = b1;
    };
    {
        h8 a0, a1, b0, b1;
        fetch(0, a0, a1, b0, b1);
        stage(0, a0, a1, b0, b1);
    }
    __syncthreads();
#endif

    #pragma unroll
    for (int kk = 0; kk < KT; ++kk) {
        const int cur = kk & 1;
#if HAVE_ASYNC
        if (kk + 1 < KT) issue(kk + 1);      // DMA into other buffer, overlapped with WMMAs
#else
        h8 na0, na1, nb0, nb1;
        if (kk + 1 < KT) fetch(kk + 1, na0, na1, nb0, nb1);
#endif

        // ---- A fragment: halves 0..7 = K akh..akh+7 ; halves 8..15 = K 16+akh..+7 ----
        const h8 alo = *(const h8*)(&As[cur][am * LDA + akh]);
        const h8 ahi = *(const h8*)(&As[cur][am * LDA + 16 + akh]);
        v16h afrag;
        #pragma unroll
        for (int i = 0; i < 8; ++i) { afrag[i] = alo[i]; afrag[8 + i] = ahi[i]; }

        #pragma unroll
        for (int nt = 0; nt < 8; ++nt) {
            const int o = nt * 16 + l15;
            const h8 blo = *(const h8*)(&Bs[cur][o * LDA + bks]);
            const h8 bhi = *(const h8*)(&Bs[cur][o * LDA + bks + 16]);
            v16h bfrag;
            #pragma unroll
            for (int i = 0; i < 8; ++i) { bfrag[i] = blo[i]; bfrag[8 + i] = bhi[i]; }
            acc[nt] = __builtin_amdgcn_wmma_f32_16x16x32_f16(
                false, afrag, false, bfrag, (short)0, acc[nt], false, false);
        }

        if (kk + 1 < KT) {
#if HAVE_ASYNC
            WAIT_ASYNC();      // this wave's DMAs into next buffer are done
            __syncthreads();   // all waves done: next buffer ready, old buffer free
#else
            stage((kk + 1) & 1, na0, na1, nb0, nb1);
            __syncthreads();
#endif
        }
    }

    // ---- quantized epilogue: D layout m_local = 8*(lane>>4)+row, n_local = lane&15 ----
    const float sc = scal[0], Th = scal[1];
    const float rTh7 = 7.0f / Th;
    float swo[8], bwo[8], bbo[8];
    #pragma unroll
    for (int nt = 0; nt < 8; ++nt) {
        const int o = nt * 16 + l15;
        swo[nt] = sw[o]; bwo[nt] = bw[o]; bbo[nt] = bb[o];
    }
    #pragma unroll
    for (int rr = 0; rr < 8; ++rr) {
        const int m = wv * 16 + lhi * 8 + rr;
        const int p = blockIdx.x * 128 + m;
        const int n = p / HW, hw = p % HW;
        const size_t resBase = (size_t)n * CC * HW + hw;   // + o*HW for NCHW access
        #pragma unroll
        for (int nt = 0; nt < 8; ++nt) {
            const int o = nt * 16 + l15;
            const float v = acc[nt][rr];
            const float aq = floorf(rintf(v * 0.25f) * swo[nt]);   // CIM accumulator quantization
            float t = rintf(aq * bwo[nt] + bbo[nt]);
            if (layer == 0) {
                const float xr = resid_nchw[resBase + (size_t)o * HW];
                t += rintf(xr * sc);
                t = fminf(fmaxf(t, -Th), Th);
                outT[(size_t)p * CC + o] = (_Float16)rintf(t * rTh7);
            } else {
                const float xr = (float)resid_nhwc[(size_t)p * CC + o];
                t = rintf(t + xr * sc);
                t = fminf(fmaxf(t, -Th), Th);
                out_nchw[resBase + (size_t)o * HW] = rintf(t * rTh7);
            }
        }
    }
}

extern "C" void kernel_launch(void* const* d_in, const int* in_sizes, int n_in,
                              void* d_out, int out_size, void* d_ws, size_t ws_size,
                              hipStream_t stream) {
    const float* x  = (const float*)d_in[0];
    const float* w1 = (const float*)d_in[1];
    const float* w2 = (const float*)d_in[2];
    const float* g1 = (const float*)d_in[3];
    const float* b1 = (const float*)d_in[4];
    const float* m1 = (const float*)d_in[5];
    const float* v1 = (const float*)d_in[6];
    const float* g2 = (const float*)d_in[7];
    const float* b2 = (const float*)d_in[8];
    const float* m2 = (const float*)d_in[9];
    const float* v2 = (const float*)d_in[10];
    float* out = (float*)d_out;

    // workspace layout
    char* ws = (char*)d_ws;
    const size_t actBytes = (size_t)NPIX * CC * sizeof(_Float16);   // 25,690,112 B
    _Float16* xT   = (_Float16*)ws;                         ws += actBytes;
    _Float16* x1T  = (_Float16*)ws;                         ws += actBytes;
    _Float16* wp   = (_Float16*)ws;                         ws += (size_t)2 * CC * KK * sizeof(_Float16);
    float*    swb  = (float*)ws;                            ws += 2 * CC * sizeof(float);
    float*    bwb  = (float*)ws;                            ws += 2 * CC * sizeof(float);
    float*    bbb  = (float*)ws;                            ws += 2 * CC * sizeof(float);
    float*    scal = (float*)ws;                            ws += 4 * sizeof(float);
    float*    zpad = (float*)ws;                            ws += 32 * sizeof(float);  // 128B zeros

    pack_w_kernel<<<dim3(256), dim3(128), 0, stream>>>(w1, w2, wp, swb);
    bn_prep_kernel<<<dim3(1), dim3(128), 0, stream>>>(g1, b1, m1, v1, g2, b2, m2, v2, bwb, bbb, scal, zpad);
    to_nhwc_kernel<<<dim3((NPIX * 32) / 256), dim3(256), 0, stream>>>(x, xT);

    // layer 1: A = xT, B = wp[0], residual = x (NCHW f32), out -> x1T (NHWC f16)
    conv_wmma_kernel<<<dim3(NPIX / 128), dim3(256), 0, stream>>>(
        xT, wp, swb, bwb, bbb, scal, (const _Float16*)zpad, x, nullptr, x1T, nullptr, 0);
    // layer 2: A = x1T, B = wp[1], residual = x1T, out -> d_out (NCHW f32)
    conv_wmma_kernel<<<dim3(NPIX / 128), dim3(256), 0, stream>>>(
        x1T, wp + (size_t)CC * KK, swb + CC, bwb + CC, bbb + CC, scal + 2,
        (const _Float16*)zpad, nullptr, x1T, nullptr, out, 1);
}
